// Atten_Agg_Layer_18262200943351
// MI455X (gfx1250) — matrix operand order repaired and checked by reading, not compiled, not run on hardware
//
#include <hip/hip_runtime.h>
#include <hip/hip_bf16.h>

#define N_NODES 50000
#define N_EDGES 800000
#define FEATS   128
#define ROWTILES (N_NODES / 16)   // 3125 exactly

typedef __attribute__((ext_vector_type(16))) __bf16 v16bf;
typedef __attribute__((ext_vector_type(8)))  float  v8f;

// ---------- helpers ----------
__device__ __forceinline__ unsigned f2bf(float f) {            // f32 -> bf16 bits, RNE
    unsigned u = __float_as_uint(f);
    return (u + 0x7FFFu + ((u >> 16) & 1u)) >> 16;
}
__device__ __forceinline__ unsigned fkey(float f) {            // monotone float -> uint
    unsigned u = __float_as_uint(f);
    return (u & 0x80000000u) ? ~u : (u | 0x80000000u);
}
__device__ __forceinline__ float kdec(unsigned k) {            // inverse of fkey
    unsigned u = (k & 0x80000000u) ? (k & 0x7FFFFFFFu) : ~k;
    return __uint_as_float(u);
}

// ---------- 1) re-init accumulators every launch (graph replay safety) ----------
__global__ void init_ws(float* __restrict__ out, unsigned* __restrict__ m,
                        float* __restrict__ s, float* __restrict__ stats) {
    int i = blockIdx.x * blockDim.x + threadIdx.x;
    const int NOUT = N_NODES * FEATS;
    const int NH   = N_NODES * 8;
    if (i < NOUT) out[i] = 0.0f;
    if (i < NH)   { m[i] = 0u; s[i] = 0.0f; }   // key 0 == smaller than any finite float
    if (i < 512)  stats[i] = 0.0f;
}

// ---------- 2) column sums / sumsq for BN (atomic partials) ----------
__global__ void col_stats(const float* __restrict__ ftq, const float* __restrict__ ftk,
                          float* __restrict__ stats) {
    const float* x = blockIdx.y ? ftk : ftq;
    float* dst = stats + blockIdx.y * 256;       // [sum 128][sumsq 128]
    int c  = threadIdx.x;                        // 0..127
    int r0 = blockIdx.x * 256;
    int r1 = min(r0 + 256, N_NODES);
    float s = 0.0f, ss = 0.0f;
    for (int r = r0; r < r1; ++r) {
        float v = x[(size_t)r * FEATS + c];
        s += v; ss += v * v;
    }
    atomicAdd(&dst[c], s);
    atomicAdd(&dst[128 + c], ss);
}

// ---------- 3) fold BN affine into weights (bf16) + folded biases ----------
__global__ void fold_weights(const float* __restrict__ Wq, const float* __restrict__ bq,
                             const float* __restrict__ Wk, const float* __restrict__ Wv,
                             const float* __restrict__ gq, const float* __restrict__ bqb,
                             const float* __restrict__ gk, const float* __restrict__ bkb,
                             const float* __restrict__ stats,
                             unsigned short* __restrict__ wqb, unsigned short* __restrict__ wkb,
                             unsigned short* __restrict__ wvb,
                             float* __restrict__ biasq, float* __restrict__ biask,
                             float* __restrict__ biasv) {
    __shared__ float aq[128], b0q[128], ak[128], b0k[128];
    int t = threadIdx.x;                         // 0..127
    {
        float mean = stats[t] * (1.0f / N_NODES);
        float var  = stats[128 + t] * (1.0f / N_NODES) - mean * mean;
        float a    = rsqrtf(var + 1e-5f) * gq[t];
        aq[t] = a; b0q[t] = bqb[t] - mean * a;
        mean = stats[256 + t] * (1.0f / N_NODES);
        var  = stats[384 + t] * (1.0f / N_NODES) - mean * mean;
        a    = rsqrtf(var + 1e-5f) * gk[t];
        ak[t] = a; b0k[t] = bkb[t] - mean * a;
    }
    __syncthreads();
    float accq = bq[t], acck = 0.0f, accv = 0.0f;
    for (int c = 0; c < 128; ++c) {
        float wq = Wq[t * 128 + c], wk = Wk[t * 128 + c], wv = Wv[t * 128 + c];
        wqb[t * 128 + c] = (unsigned short)f2bf(wq * aq[c]);
        wkb[t * 128 + c] = (unsigned short)f2bf(wk * ak[c]);
        wvb[t * 128 + c] = (unsigned short)f2bf(wv * aq[c]);
        accq += wq * b0q[c];
        acck += wk * b0k[c];
        accv += wv * b0q[c];
    }
    biasq[t] = accq; biask[t] = acck; biasv[t] = accv;
}

// ---------- 4) raw features -> bf16 copies ----------
__global__ void cvt_bf16(const float* __restrict__ ftq, const float* __restrict__ ftk,
                         unsigned short* __restrict__ xq, unsigned short* __restrict__ xk) {
    int i = (blockIdx.x * blockDim.x + threadIdx.x) * 4;
    const int N = N_NODES * FEATS;
    if (i >= N) return;
    float4 a = *(const float4*)(ftq + i);
    float4 b = *(const float4*)(ftk + i);
    uint2 oa, ob;
    oa.x = f2bf(a.x) | (f2bf(a.y) << 16);
    oa.y = f2bf(a.z) | (f2bf(a.w) << 16);
    ob.x = f2bf(b.x) | (f2bf(b.y) << 16);
    ob.y = f2bf(b.z) | (f2bf(b.w) << 16);
    *(uint2*)(xq + i) = oa;
    *(uint2*)(xk + i) = ob;
}

// ---------- 5) q/k/v GEMM via bf16 WMMA; one wave = 16 rows x 128 cols ----------
__global__ void gemm_qkv(const unsigned short* __restrict__ xq,
                         const unsigned short* __restrict__ xk,
                         const unsigned short* __restrict__ wqb,
                         const unsigned short* __restrict__ wkb,
                         const unsigned short* __restrict__ wvb,
                         const float* __restrict__ biasq, const float* __restrict__ biask,
                         const float* __restrict__ biasv,
                         float* __restrict__ qo, float* __restrict__ ko,
                         float* __restrict__ vo) {
    const int wave = threadIdx.x >> 5;
    const int lane = threadIdx.x & 31;
    const int mt   = blockIdx.x * 4 + wave;            // row tile
    if (mt >= ROWTILES) return;                        // wave-uniform: EXEC stays all-ones
    const int which = blockIdx.y;                      // 0=q 1=k 2=v
    const unsigned short* X = (which == 1) ? xk : xq;
    const unsigned short* W = (which == 0) ? wqb : (which == 1) ? wkb : wvb;
    const float* bias = (which == 0) ? biasq : (which == 1) ? biask : biasv;
    float* out = (which == 0) ? qo : (which == 1) ? ko : vo;

    const int row0 = mt * 16;
    const int half = lane >> 4;                        // 0/1
    const int lm   = lane & 15;

    // A fragments: lane = row lm; half 0 -> K {0..7,16..23}, half 1 -> K {8..15,24..31}
    v16bf afrag[4];
    const unsigned short* arow = X + (size_t)(row0 + lm) * FEATS;
#pragma unroll
    for (int c = 0; c < 4; ++c) {
        int kb = c * 32 + half * 8;
        union { v16bf bf; uint4 u[2]; } t;
        t.u[0] = *(const uint4*)(arow + kb);
        t.u[1] = *(const uint4*)(arow + kb + 16);
        afrag[c] = t.bf;
    }
#pragma unroll
    for (int nt = 0; nt < 8; ++nt) {
        const int n0 = nt * 16;
        const unsigned short* brow = W + (size_t)(n0 + lm) * FEATS;   // B[k,n] = W'[n,k]
        v8f acc = {};
#pragma unroll
        for (int c = 0; c < 4; ++c) {
            int kb = c * 32 + half * 8;                // K {0..7,16..23} / {8..15,24..31} per half
            union { v16bf bf; uint4 u[2]; } t;
            t.u[0] = *(const uint4*)(brow + kb);
            t.u[1] = *(const uint4*)(brow + kb + 16);
            acc = __builtin_amdgcn_wmma_f32_16x16x32_bf16(
                false, afrag[c], false, t.bf, (short)0, acc, false, false);
        }
        float b = bias[n0 + lm];
#pragma unroll
        for (int r = 0; r < 8; ++r)                     // C: row r+half*8, col lm
            out[(size_t)(row0 + r + half * 8) * FEATS + n0 + lm] = acc[r] + b;
    }
}

// ---------- 6) per-edge logits + segment max (wave per edge) ----------
__global__ void edge_logits(const float* __restrict__ q, const float* __restrict__ k,
                            const int* __restrict__ src, const int* __restrict__ dst,
                            const float* __restrict__ attn_e,
                            float* __restrict__ e, unsigned* __restrict__ m) {
    int gtid = blockIdx.x * blockDim.x + threadIdx.x;
    int wid  = gtid >> 5;
    int lane = threadIdx.x & 31;
    if (wid >= N_EDGES) return;
    int sn = src[wid], dn = dst[wid];
    float4 qv = *(const float4*)(q + (size_t)sn * FEATS + lane * 4);
    float4 kv = *(const float4*)(k + (size_t)dn * FEATS + lane * 4);
    float4 ae = *(const float4*)(attn_e + lane * 4);
    float t = ae.x / (1.0f + __expf(-(qv.x + kv.x)))
            + ae.y / (1.0f + __expf(-(qv.y + kv.y)))
            + ae.z / (1.0f + __expf(-(qv.z + kv.z)))
            + ae.w / (1.0f + __expf(-(qv.w + kv.w)));
    t += __shfl_xor(t, 1, 32);                         // reduce 4-lane head group
    t += __shfl_xor(t, 2, 32);
    int h = lane >> 2;
    if ((lane & 3) == 0) {
        e[(size_t)wid * 8 + h] = t;
        atomicMax(&m[(size_t)dn * 8 + h], fkey(t));
    }
}

// ---------- 7) exp + segment sum (thread per edge*head) ----------
__global__ void edge_exp(const int* __restrict__ dst, float* __restrict__ e,
                         const unsigned* __restrict__ m, float* __restrict__ s) {
    int i = blockIdx.x * blockDim.x + threadIdx.x;
    if (i >= N_EDGES * 8) return;
    int edge = i >> 3, h = i & 7;
    int dn = dst[edge];
    float ex = __expf(e[i] - kdec(m[(size_t)dn * 8 + h]));
    e[i] = ex;                                          // reuse buffer
    atomicAdd(&s[(size_t)dn * 8 + h], ex);
}

// ---------- 8) normalize + weighted scatter-add of v[src] (wave per edge) ----------
__global__ void edge_agg(const float* __restrict__ v, const int* __restrict__ src,
                         const int* __restrict__ dst, const float* __restrict__ ex,
                         const float* __restrict__ s, float* __restrict__ out) {
    int gtid = blockIdx.x * blockDim.x + threadIdx.x;
    int wid  = gtid >> 5;
    int lane = threadIdx.x & 31;
    if (wid >= N_EDGES) return;
    int sn = src[wid], dn = dst[wid];
    int h = lane >> 2;
    float a = ex[(size_t)wid * 8 + h] / s[(size_t)dn * 8 + h];
    float4 vv = *(const float4*)(v + (size_t)sn * FEATS + lane * 4);
    float* op = out + (size_t)dn * FEATS + lane * 4;
    atomicAdd(op + 0, vv.x * a);
    atomicAdd(op + 1, vv.y * a);
    atomicAdd(op + 2, vv.z * a);
    atomicAdd(op + 3, vv.w * a);
}

extern "C" void kernel_launch(void* const* d_in, const int* in_sizes, int n_in,
                              void* d_out, int out_size, void* d_ws, size_t ws_size,
                              hipStream_t stream) {
    const float* ft_q   = (const float*)d_in[0];
    const float* ft_k   = (const float*)d_in[1];
    const int*   src    = (const int*)d_in[2];
    const int*   dst    = (const int*)d_in[3];
    const float* Wq     = (const float*)d_in[4];
    const float* bq     = (const float*)d_in[5];
    const float* Wk     = (const float*)d_in[6];
    const float* Wv     = (const float*)d_in[7];
    const float* attn_e = (const float*)d_in[8];
    const float* bn_q_g = (const float*)d_in[9];
    const float* bn_q_b = (const float*)d_in[10];
    const float* bn_k_g = (const float*)d_in[11];
    const float* bn_k_b = (const float*)d_in[12];
    float* out = (float*)d_out;

    const size_t NF = (size_t)N_NODES * FEATS;          // 6,400,000
    const size_t NE8 = (size_t)N_EDGES * 8;             // 6,400,000
    const size_t NH  = (size_t)N_NODES * 8;             //   400,000

    float* q = (float*)d_ws;
    float* k = q + NF;
    float* v = k + NF;
    float* e = v + NF;                                  // NE8 floats (reused as ex)
    unsigned* m = (unsigned*)(e + NE8);                 // NH keys
    float* s = (float*)(m + NH);                        // NH floats
    unsigned short* xq  = (unsigned short*)(s + NH);    // NF bf16
    unsigned short* xk  = xq + NF;
    unsigned short* wqb = xk + NF;                      // 128*128 bf16 each
    unsigned short* wkb = wqb + 16384;
    unsigned short* wvb = wkb + 16384;
    float* biasq = (float*)(wvb + 16384);
    float* biask = biasq + 128;
    float* biasv = biask + 128;
    float* stats = biasv + 128;                         // 512 floats

    // 1) init accumulators (also zeroes d_out)
    init_ws<<<(int)((NF + 255) / 256), 256, 0, stream>>>(out, m, s, stats);
    // 2) BN column stats
    col_stats<<<dim3(196, 2), 128, 0, stream>>>(ft_q, ft_k, stats);
    // 3) fold BN into bf16 weights + biases
    fold_weights<<<1, 128, 0, stream>>>(Wq, bq, Wk, Wv, bn_q_g, bn_q_b, bn_k_g, bn_k_b,
                                        stats, wqb, wkb, wvb, biasq, biask, biasv);
    // 4) raw features -> bf16
    cvt_bf16<<<(int)((NF / 4 + 255) / 256), 256, 0, stream>>>(ft_q, ft_k, xq, xk);
    // 5) WMMA GEMMs for q, k, v
    gemm_qkv<<<dim3((ROWTILES + 3) / 4, 3), 128, 0, stream>>>(
        xq, xk, wqb, wkb, wvb, biasq, biask, biasv, q, k, v);
    // 6) edge logits + segment max
    edge_logits<<<(int)(((size_t)N_EDGES * 32 + 255) / 256), 256, 0, stream>>>(
        q, k, src, dst, attn_e, e, m);
    // 7) exp + segment sum
    edge_exp<<<(int)((NE8 + 255) / 256), 256, 0, stream>>>(dst, e, m, s);
    // 8) normalize + scatter-add aggregation
    edge_agg<<<(int)(((size_t)N_EDGES * 32 + 255) / 256), 256, 0, stream>>>(
        v, src, dst, e, s, out);
}
